// FABlockS2_9234179686827
// MI455X (gfx1250) — matrix operand order repaired and checked by reading, not compiled, not run on hardware
//
#include <hip/hip_runtime.h>
#include <hip/hip_bf16.h>

typedef __bf16 bf16;
typedef __attribute__((ext_vector_type(16))) __bf16 v16bf;
typedef __attribute__((ext_vector_type(8)))  float  v8f;

__device__ __forceinline__ float gelu_f(float x) {
  const float c0 = 0.7978845608028654f; // sqrt(2/pi)
  return 0.5f * x * (1.0f + tanhf(c0 * (x + 0.044715f * x * x * x)));
}

// ---------------------------------------------------------------------------
// Generic batched WMMA GEMM: C = alpha * A(MxK) x B(KxN) (+bias) (+epilogue)
// A/B staged to LDS as bf16; accumulate fp32 via v_wmma_f32_16x16x32_bf16.
// bf16 A-tiles are staged with global_load_async_to_lds_b128 (ASYNCcnt path);
// f32 operands go through VGPRs for the f32->bf16 convert.
// Tiles: BM=64 BN=64 BK=64, 256 threads = 8 waves, wave -> 16x32 of C.
// Batch z decomposed as (b = z/HP, h = z%HP) with independent strides.
// EPI:     0 none | 1 gelu | 2 (*mod, *jw?, leaky 0.2) | 3 (+skip)
// OUTMODE: 0 f32 linear | 1 bf16 linear | 2 bf16 scatter-V | 3 bf16 scatter-T1
//          | 4 f32 scatter-UPHI
// ---------------------------------------------------------------------------
struct GemmParams {
  const void* A; const void* B; const float* bias; void* C;
  const float* mod; const float* jw; const float* skip;
  long lda, ldb, ldc;
  long sAb, sAh, sBb, sBh, sCb, sCh, sModh;
  int M, N, K, HP;
  float alpha;
};

template<typename AT, typename BT, bool TRANSB, int EPI, int OUTMODE>
__global__ __launch_bounds__(256) void gemm_wmma(GemmParams p) {
  constexpr int BK = 64;
  constexpr int APAD = 8;                       // 72 elems = 144B rows, 16B aligned
  constexpr bool kAsyncA = __is_same(AT, bf16); // raw bf16 bytes -> async DMA to LDS
  __shared__ bf16 Al[64][BK + APAD];
  __shared__ bf16 Bl[64][BK + 2];
  const int z = blockIdx.z;
  const int b = z / p.HP, h = z % p.HP;
  const AT* Ag = (const AT*)p.A + (size_t)b * p.sAb + (size_t)h * p.sAh;
  const BT* Bg = (const BT*)p.B + (size_t)b * p.sBb + (size_t)h * p.sBh;
  const int m0 = blockIdx.y * 64, n0 = blockIdx.x * 64;
  const int tid = threadIdx.x, lane = tid & 31, wv = tid >> 5;
  const int wm = wv >> 1, wn = wv & 1, hlf = lane >> 4, l15 = lane & 15;
  v8f accs[2] = {};
  for (int kt = 0; kt < p.K; kt += BK) {
    if constexpr (kAsyncA) {
      // 64x64 bf16 tile = 512 x 16B chunks; 8 chunks per row.
#pragma unroll
      for (int i = 0; i < 2; ++i) {
        int e = tid + i * 256;
        int mm = e >> 3, c16 = e & 7;
        unsigned lds = (unsigned)(unsigned long long)(void*)&Al[mm][c16 * 8];
        unsigned long long ga = (unsigned long long)(const void*)
            (Ag + (size_t)(m0 + mm) * (size_t)p.lda + kt + c16 * 8);
        asm volatile("global_load_async_to_lds_b128 %0, %1, off"
                     :: "v"(lds), "v"(ga) : "memory");
      }
    } else {
#pragma unroll
      for (int i = 0; i < 16; ++i) {
        int e = tid + i * 256;
        int mm = e >> 6, kk = e & 63;
        Al[mm][kk] = (bf16)(float)Ag[(size_t)(m0 + mm) * (size_t)p.lda + (kt + kk)];
      }
    }
#pragma unroll
    for (int i = 0; i < 16; ++i) {
      int e = tid + i * 256;
      int nn = e >> 6, kk = e & 63;
      float v;
      if (TRANSB) v = (float)Bg[(size_t)(n0 + nn) * (size_t)p.ldb + (kt + kk)];
      else        v = (float)Bg[(size_t)(kt + kk) * (size_t)p.ldb + (n0 + nn)];
      Bl[nn][kk] = (bf16)v;
    }
    if constexpr (kAsyncA) asm volatile("s_wait_asynccnt 0x0" ::: "memory");
    __syncthreads();
#pragma unroll
    for (int ks = 0; ks < BK; ks += 32) {
      v16bf a, b0, b1;
      const int ar = wm * 16 + l15;
#pragma unroll
      for (int e = 0; e < 8; ++e) {
        a[e]     = Al[ar][ks + hlf * 8 + e];        // K = half*8 + e
        a[8 + e] = Al[ar][ks + 16 + hlf * 8 + e];   // K = 16 + half*8 + e
      }
      const int bc = wn * 32 + l15;
#pragma unroll
      for (int e = 0; e < 8; ++e) {
        b0[e]     = Bl[bc][ks + hlf * 8 + e];          // K = half*8 + e
        b0[8 + e] = Bl[bc][ks + 16 + hlf * 8 + e];     // K = 16 + half*8 + e
        b1[e]     = Bl[bc + 16][ks + hlf * 8 + e];
        b1[8 + e] = Bl[bc + 16][ks + 16 + hlf * 8 + e];
      }
      accs[0] = __builtin_amdgcn_wmma_f32_16x16x32_bf16(false, a, false, b0, (short)0, accs[0], false, false);
      accs[1] = __builtin_amdgcn_wmma_f32_16x16x32_bf16(false, a, false, b1, (short)0, accs[1], false, false);
    }
    __syncthreads();
  }
  const float* modp = (EPI == 2) ? (p.mod + (size_t)h * p.sModh) : nullptr;
#pragma unroll
  for (int part = 0; part < 2; ++part) {
    const int gc = n0 + wn * 32 + part * 16 + l15;  // N = lane&15
    const float bv = p.bias ? p.bias[gc] : 0.0f;    // hoisted: gc invariant over r
#pragma unroll
    for (int r = 0; r < 8; ++r) {
      const int gr = m0 + wm * 16 + r + hlf * 8;    // C: vgpr r, lanes16-31 -> M+8
      float v = accs[part][r] * p.alpha + bv;
      if (EPI == 1) v = gelu_f(v);
      if (EPI == 2) {
        v *= modp[(size_t)gr * p.N + gc];
        if (p.jw) v *= p.jw[gc];
        v = (v > 0.0f) ? v : 0.2f * v;
      }
      if (EPI == 3) v += p.skip[(size_t)gr * p.N + gc];
      size_t idx;
      if (OUTMODE == 0 || OUTMODE == 1) {
        idx = (size_t)b * p.sCb + (size_t)h * p.sCh + (size_t)gr * p.ldc + gc;
      } else if (OUTMODE == 2) { // row=(b,lat,lon), col=(h,c) -> (b,h,lat,lon,c)
        int bb = gr >> 15, rem = gr & 32767, la = rem >> 8, lo = rem & 255;
        idx = ((((size_t)bb * 8 + (gc >> 6)) * 128 + la) * 256 + lo) * 64 + (gc & 63);
      } else if (OUTMODE == 3) { // row=i(lat), col=(m,c) -> (z, m, i, c)
        idx = ((size_t)z * 256 + (gc >> 6)) * 8192 + (size_t)gr * 64 + (gc & 63);
      } else {                   // row=l(lon), col=(i,c) -> (b, i, l, h, c)
        idx = (((size_t)b * 128 + (gc >> 6)) * 256 + gr) * 512 + (size_t)h * 64 + (gc & 63);
      }
      if (OUTMODE == 0 || OUTMODE == 4) ((float*)p.C)[idx] = v;
      else ((bf16*)p.C)[idx] = (bf16)v;
    }
  }
}

// ------------------------- small support kernels ---------------------------
__global__ void pool_lon_kernel(const bf16* uu, float* out) { // mean over lat
  int t = blockIdx.x * blockDim.x + threadIdx.x; // B*NLON*DIM = 131072
  if (t >= 2 * 256 * 256) return;
  int d = t & 255, lo = (t >> 8) & 255, b = t >> 16;
  const bf16* base = uu + ((size_t)b * 32768 + lo) * 256 + d;
  float s = 0.f;
  for (int la = 0; la < 128; ++la) s += (float)base[(size_t)la * 65536];
  out[t] = s * (1.0f / 128.0f);
}

__global__ void pool_lat_kernel(const bf16* uu, float* out) { // mean over lon
  int t = blockIdx.x * blockDim.x + threadIdx.x; // B*NLAT*DIM = 65536
  if (t >= 2 * 128 * 256) return;
  int d = t & 255, la = (t >> 8) & 127, b = t >> 15;
  const bf16* base = uu + ((size_t)b * 32768 + (size_t)la * 256) * 256 + d;
  float s = 0.f;
  for (int lo = 0; lo < 256; ++lo) s += (float)base[(size_t)lo * 256];
  out[t] = s * (1.0f / 256.0f);
}

__global__ void wlat_kernel(const float* lat, float* w) {
  int i = threadIdx.x; // 128
  float dlat = lat[1] - lat[0];
  float sq = sinf(dlat * 0.25f);
  float pw = 1.0f * sq * sq / sinf(dlat * 0.5f); // POLAR_EPS = 1
  float v = cosf(lat[i]);
  if ((i == 0 || i == 127) && v < 0.001f) v = pw;
  w[i] = v;
}

// fused: y=(x@in_w)*scale; LN; h=y@fc1+b; glu; out=glu@fc2+b. One block per row.
__global__ __launch_bounds__(256) void pool_head_kernel(
    const float* pooled, const float* in_w, const float* ln_g, const float* ln_b,
    const float* fc1_w, const float* fc1_b, const float* fc2_w, const float* fc2_b,
    const float* rowscale, int nrows_per_b, float* out) {
  __shared__ float x[256], y[512], red[256];
  int row = blockIdx.x, t = threadIdx.x;
  x[t] = pooled[(size_t)row * 256 + t];
  __syncthreads();
  float acc = 0.f;
  for (int k = 0; k < 256; ++k) acc += x[k] * in_w[k * 256 + t];
  if (rowscale) acc *= rowscale[row % nrows_per_b];
  y[t] = acc; red[t] = acc;
  __syncthreads();
  for (int s = 128; s > 0; s >>= 1) { if (t < s) red[t] += red[t + s]; __syncthreads(); }
  float mean = red[0] * (1.0f / 256.0f);
  __syncthreads();
  float d = y[t] - mean;
  red[t] = d * d;
  __syncthreads();
  for (int s = 128; s > 0; s >>= 1) { if (t < s) red[t] += red[t + s]; __syncthreads(); }
  float var = red[0] * (1.0f / 256.0f);
  __syncthreads();
  x[t] = d * rsqrtf(var + 1e-5f) * ln_g[t] + ln_b[t];
  __syncthreads();
  for (int j = t; j < 512; j += 256) {
    float a2 = fc1_b[j];
    for (int k = 0; k < 256; ++k) a2 += x[k] * fc1_w[k * 512 + j];
    y[j] = a2;
  }
  __syncthreads();
  red[t] = y[t] * gelu_f(y[t + 256]);
  __syncthreads();
  float o = fc2_b[t];
  for (int k = 0; k < 256; ++k) o += red[k] * fc2_w[k * 256 + t];
  out[(size_t)row * 256 + t] = o;
}

__global__ void bessel_kernel(const float* coord, int n, const float* freq, int nf,
                              const float* w, int periodic, float* out) {
  int t = blockIdx.x * blockDim.x + threadIdx.x;
  if (t >= 8 * n * n) return;
  int j = t % n, i = (t / n) % n, h = t / (n * n);
  float d = fabsf(coord[i] - coord[j]);
  if (periodic) d = fminf(d, 6.283185307179586f - d);
  float acc = 0.f;
  if (d > 1e-6f) {
    float inv = 1.0f / d;
    for (int k = 0; k < nf; ++k) acc += sinf(freq[k] * d) * inv * w[k * 8 + h];
  } else {
    for (int k = 0; k < nf; ++k) acc += freq[k] * w[k * 8 + h];
  }
  out[((size_t)h * n + i) * n + j] = acc;
}

__global__ void gn_zero_kernel(float* stats) { if (threadIdx.x < 32) stats[threadIdx.x] = 0.f; }

__global__ __launch_bounds__(256) void gn_stats_kernel(const float* uphi, float* stats) {
  int g = blockIdx.y, b = g >> 3, h = g & 7;
  size_t base = (size_t)blockIdx.x * 1024 + (size_t)threadIdx.x * 4;
  float s = 0.f, sq = 0.f;
  for (int q = 0; q < 4; ++q) {
    size_t e = base + q;                 // over 2,097,152 elems of group
    size_t sp = e >> 6; int c = (int)(e & 63);
    float v = uphi[((size_t)b * 32768 + sp) * 512 + h * 64 + c];
    s += v; sq += v * v;
  }
  __shared__ float rs[256], rq[256];
  rs[threadIdx.x] = s; rq[threadIdx.x] = sq;
  __syncthreads();
  for (int st = 128; st > 0; st >>= 1) {
    if (threadIdx.x < st) { rs[threadIdx.x] += rs[threadIdx.x + st]; rq[threadIdx.x] += rq[threadIdx.x + st]; }
    __syncthreads();
  }
  if (threadIdx.x == 0) { atomicAdd(&stats[2 * g], rs[0]); atomicAdd(&stats[2 * g + 1], rq[0]); }
}

__global__ void gn_apply_kernel(const float* uphi, const float* stats,
                                const float* g, const float* bb, bf16* out) {
  size_t t = (size_t)blockIdx.x * 256 + threadIdx.x;
  if (t >= (size_t)65536 * 512) return;
  int c512 = (int)(t & 511);
  size_t sp = t >> 9;
  int b = (int)(sp >> 15), h = c512 >> 6, grp = b * 8 + h;
  const float inv = 1.0f / 2097152.0f;
  float m = stats[2 * grp] * inv;
  float var = stats[2 * grp + 1] * inv - m * m;
  float v = (uphi[t] - m) * rsqrtf(var + 1e-5f);
  out[t] = (bf16)(v * g[c512] + bb[c512]);
}

// ------------------------------- launcher ----------------------------------
extern "C" void kernel_launch(void* const* d_in, const int* in_sizes, int n_in,
                              void* d_out, int out_size, void* d_ws, size_t ws_size,
                              hipStream_t stream) {
  (void)in_sizes; (void)n_in; (void)out_size; (void)ws_size;
  const float* u          = (const float*)d_in[0];
  const float* lat        = (const float*)d_in[1];
  const float* lon        = (const float*)d_in[2];
  const float* cm_w1      = (const float*)d_in[3];
  const float* cm_b1      = (const float*)d_in[4];
  const float* cm_w2      = (const float*)d_in[5];
  const float* cm_b2      = (const float*)d_in[6];
  const float* long_in_w  = (const float*)d_in[7];
  const float* long_ln_g  = (const float*)d_in[8];
  const float* long_ln_b  = (const float*)d_in[9];
  const float* long_fc1_w = (const float*)d_in[10];
  const float* long_fc1_b = (const float*)d_in[11];
  const float* long_fc2_w = (const float*)d_in[12];
  const float* long_fc2_b = (const float*)d_in[13];
  const float* lat_in_w   = (const float*)d_in[14];
  const float* lat_ln_g   = (const float*)d_in[15];
  const float* lat_ln_b   = (const float*)d_in[16];
  const float* lat_fc1_w  = (const float*)d_in[17];
  const float* lat_fc1_b  = (const float*)d_in[18];
  const float* lat_fc2_w  = (const float*)d_in[19];
  const float* lat_fc2_b  = (const float*)d_in[20];
  const float* klong_q_w  = (const float*)d_in[21];
  const float* klong_k_w  = (const float*)d_in[22];
  const float* klat_q_w   = (const float*)d_in[23];
  const float* klat_k_w   = (const float*)d_in[24];
  const float* pe_long_fr = (const float*)d_in[25];
  const float* pe_long_w  = (const float*)d_in[26];
  const float* pe_lat_fr  = (const float*)d_in[27];
  const float* pe_lat_w   = (const float*)d_in[28];
  const float* gn_g       = (const float*)d_in[29];
  const float* gn_b       = (const float*)d_in[30];
  const float* merge_w    = (const float*)d_in[31];
  const float* out_w      = (const float*)d_in[32];

  // ---- workspace layout (bytes), with region reuse ----
  char* ws = (char*)d_ws;
  bf16*  h1    = (bf16*)(ws + 0);                       // 65536x1536 bf16 (201326592B)
  bf16*  t1    = (bf16*)(ws + 0);                       // reuse: (b,h,m,i,c) bf16 (67108864B)
  float* uphi  = (float*)(ws + 67108864);               // reuse: (b,i,l,h,c) f32 (134217728B)
  bf16*  norm  = (bf16*)(ws + 0);                       // reuse: 65536x512 bf16
  const size_t base2 = 201326592;
  bf16*  vperm = (bf16*)(ws + base2);                   // (b,h,lat,lon,c) bf16 (67108864B)
  bf16*  uu    = (bf16*)(ws + base2 + 67108864);        // 65536x256 bf16 (33554432B)
  bf16*  merged= uu;                                    // reuse after pooling
  float* uskip = (float*)(ws + base2 + 100663296);      // 65536x256 f32 (67108864B)
  const size_t s0 = base2 + 167772160;
  float* pool_lon_buf = (float*)(ws + s0 + 0);          // 524288B
  float* pool_lat_buf = (float*)(ws + s0 + 524288);     // 262144B
  float* u_long_buf   = (float*)(ws + s0 + 786432);     // 524288B
  float* u_lat_buf    = (float*)(ws + s0 + 1310720);    // 262144B
  float* wlat_buf     = (float*)(ws + s0 + 1572864);    // 1024B
  float* mod_long_buf = (float*)(ws + s0 + 1573888);    // 2097152B
  float* mod_lat_buf  = (float*)(ws + s0 + 3671040);    // 524288B
  bf16*  q_long       = (bf16*)(ws + s0 + 4195328);     // 1572864B
  bf16*  k_long_p     = (bf16*)(ws + s0 + 5768192);     // 1572864B
  bf16*  q_lat        = (bf16*)(ws + s0 + 7341056);     // 786432B
  bf16*  k_lat_p      = (bf16*)(ws + s0 + 8127488);     // 786432B
  bf16*  k_long_bf    = (bf16*)(ws + s0 + 8913920);     // 2097152B
  bf16*  k_lat_bf     = (bf16*)(ws + s0 + 11011072);    // 524288B
  float* gnstats      = (float*)(ws + s0 + 11535360);   // 128B

  auto mk = [](const void* A, long lda, const void* B, long ldb, const float* bias,
               void* C, long ldc, int M, int N, int K) {
    GemmParams p{}; p.A = A; p.B = B; p.bias = bias; p.C = C;
    p.lda = lda; p.ldb = ldb; p.ldc = ldc; p.M = M; p.N = N; p.K = K;
    p.HP = 1; p.alpha = 1.0f;
    return p;
  };
  dim3 blk(256);
  const float PI = 3.14159265358979323846f;

  // 1) h1 = gelu(u @ cm_w1 + b1)        [65536x1536, K=256]
  { GemmParams p = mk(u, 256, cm_w1, 1536, cm_b1, h1, 1536, 65536, 1536, 256);
    gemm_wmma<float, float, false, 1, 1><<<dim3(24, 1024, 1), blk, 0, stream>>>(p); }
  // 2) split h2 = h1 @ cm_w2 + b2 into v (permuted bf16), uu (bf16), u_skip (f32)
  { GemmParams p = mk(h1, 1536, cm_w2, 1024, cm_b2, vperm, 0, 65536, 512, 1536);
    gemm_wmma<bf16, float, false, 0, 2><<<dim3(8, 1024, 1), blk, 0, stream>>>(p); }
  { GemmParams p = mk(h1, 1536, cm_w2 + 512, 1024, cm_b2 + 512, uu, 256, 65536, 256, 1536);
    gemm_wmma<bf16, float, false, 0, 1><<<dim3(4, 1024, 1), blk, 0, stream>>>(p); }
  { GemmParams p = mk(h1, 1536, cm_w2 + 768, 1024, cm_b2 + 768, uskip, 256, 65536, 256, 1536);
    gemm_wmma<bf16, float, false, 0, 0><<<dim3(4, 1024, 1), blk, 0, stream>>>(p); }
  // 3) pooled means (commuted with linear layers) + quadrature weights
  pool_lon_kernel<<<512, 256, 0, stream>>>(uu, pool_lon_buf);
  pool_lat_kernel<<<256, 256, 0, stream>>>(uu, pool_lat_buf);
  wlat_kernel<<<1, 128, 0, stream>>>(lat, wlat_buf);
  // 4) pool heads -> u_long (B*256 x 256), u_lat (B*128 x 256)
  pool_head_kernel<<<512, 256, 0, stream>>>(pool_lon_buf, long_in_w, long_ln_g, long_ln_b,
      long_fc1_w, long_fc1_b, long_fc2_w, long_fc2_b, nullptr, 256, u_long_buf);
  pool_head_kernel<<<256, 256, 0, stream>>>(pool_lat_buf, lat_in_w, lat_ln_g, lat_ln_b,
      lat_fc1_w, lat_fc1_b, lat_fc2_w, lat_fc2_b, wlat_buf, 128, u_lat_buf);
  // 5) Bessel positional modulations
  bessel_kernel<<<2048, 256, 0, stream>>>(lon, 256, pe_long_fr, 64, pe_long_w, 1, mod_long_buf);
  bessel_kernel<<<512, 256, 0, stream>>>(lat, 128, pe_lat_fr, 32, pe_lat_w, 0, mod_lat_buf);
  // 6) q/k projections (store bf16)
  { GemmParams p = mk(u_long_buf, 256, klong_q_w, 1536, nullptr, q_long, 1536, 512, 1536, 256);
    gemm_wmma<float, float, false, 0, 1><<<dim3(24, 8, 1), blk, 0, stream>>>(p); }
  { GemmParams p = mk(u_long_buf, 256, klong_k_w, 1536, nullptr, k_long_p, 1536, 512, 1536, 256);
    gemm_wmma<float, float, false, 0, 1><<<dim3(24, 8, 1), blk, 0, stream>>>(p); }
  { GemmParams p = mk(u_lat_buf, 256, klat_q_w, 1536, nullptr, q_lat, 1536, 256, 1536, 256);
    gemm_wmma<float, float, false, 0, 1><<<dim3(24, 4, 1), blk, 0, stream>>>(p); }
  { GemmParams p = mk(u_lat_buf, 256, klat_k_w, 1536, nullptr, k_lat_p, 1536, 256, 1536, 256);
    gemm_wmma<float, float, false, 0, 1><<<dim3(24, 4, 1), blk, 0, stream>>>(p); }
  // 7) K_long = leaky(q k^T * mod)  [batch (b,h)=16, 256x256, K=192]
  { GemmParams p = mk(q_long, 1536, k_long_p, 1536, nullptr, k_long_bf, 256, 256, 256, 192);
    p.HP = 8; p.sAb = 256L * 1536; p.sAh = 192; p.sBb = 256L * 1536; p.sBh = 192;
    p.sCb = 8L * 65536; p.sCh = 65536; p.mod = mod_long_buf; p.sModh = 65536;
    gemm_wmma<bf16, bf16, true, 2, 1><<<dim3(4, 4, 16), blk, 0, stream>>>(p); }
  //    K_lat = leaky(q k^T * mod) * wlat_j  [batch 16, 128x128, K=192]
  { GemmParams p = mk(q_lat, 1536, k_lat_p, 1536, nullptr, k_lat_bf, 128, 128, 128, 192);
    p.HP = 8; p.sAb = 128L * 1536; p.sAh = 192; p.sBb = 128L * 1536; p.sBh = 192;
    p.sCb = 8L * 16384; p.sCh = 16384; p.mod = mod_lat_buf; p.sModh = 16384; p.jw = wlat_buf;
    gemm_wmma<bf16, bf16, true, 2, 1><<<dim3(2, 2, 16), blk, 0, stream>>>(p); }
  // 8) einsum over latitude: t1(b,h,m,i,c) = pi/128 * K_lat @ vperm  [16 x (128x16384, K=128)]
  { GemmParams p = mk(k_lat_bf, 128, vperm, 16384, nullptr, t1, 0, 128, 16384, 128);
    p.HP = 8; p.sAb = 8L * 16384; p.sAh = 16384; p.sBb = 16777216L; p.sBh = 2097152L;
    p.alpha = PI / 128.0f;
    gemm_wmma<bf16, bf16, false, 0, 3><<<dim3(256, 2, 16), blk, 0, stream>>>(p); }
  // 9) einsum over longitude: uphi(b,i,l,h,c) = 2pi/256 * K_long @ t1  [16 x (256x8192, K=256)]
  { GemmParams p = mk(k_long_bf, 256, t1, 8192, nullptr, uphi, 0, 256, 8192, 256);
    p.HP = 8; p.sAb = 8L * 65536; p.sAh = 65536; p.sBb = 16777216L; p.sBh = 2097152L;
    p.alpha = 2.0f * PI / 256.0f;
    gemm_wmma<bf16, bf16, false, 0, 4><<<dim3(128, 4, 16), blk, 0, stream>>>(p); }
  // 10) group norm (16 groups over 2M elems each), normalized -> bf16
  gn_zero_kernel<<<1, 32, 0, stream>>>(gnstats);
  gn_stats_kernel<<<dim3(2048, 16), 256, 0, stream>>>(uphi, gnstats);
  gn_apply_kernel<<<131072, 256, 0, stream>>>(uphi, gnstats, gn_g, gn_b, norm);
  // 11) merged = norm @ merge_w + u_skip   [65536x256, K=512]
  { GemmParams p = mk(norm, 512, merge_w, 256, nullptr, merged, 256, 65536, 256, 512);
    p.skip = uskip;
    gemm_wmma<bf16, float, false, 3, 1><<<dim3(4, 1024, 1), blk, 0, stream>>>(p); }
  // 12) out = merged @ out_w               [65536x256, K=256] -> f32 d_out
  { GemmParams p = mk(merged, 256, out_w, 256, nullptr, d_out, 256, 65536, 256, 256);
    gemm_wmma<bf16, float, false, 0, 0><<<dim3(4, 1024, 1), blk, 0, stream>>>(p); }
}
